// Attention_59854664237613
// MI455X (gfx1250) — matrix operand order repaired and checked by reading, not compiled, not run on hardware
//
#include <hip/hip_runtime.h>
#include <hip/hip_bf16.h>

// Sizes fixed by the reference.
#define BB 8
#define QL 2048
#define KL 2048
#define DD 1024
#define NEG_BIG (-1e12f)

typedef __attribute__((ext_vector_type(16))) __bf16 v16bf;
typedef __attribute__((ext_vector_type(8)))  __bf16 v8bf;
typedef __attribute__((ext_vector_type(8)))  float  v8f;

union V16 { v16bf v; v8bf h[2]; };
union BF2 { __bf16 h[2]; unsigned u; };

// ---------------------------------------------------------------------------
// fp32 -> bf16 row-major convert (Q and K). 8 elements / thread.
// ---------------------------------------------------------------------------
__global__ void cvt_bf16_kernel(const float* __restrict__ src,
                                __bf16* __restrict__ dst) {
  size_t i = ((size_t)blockIdx.x * blockDim.x + threadIdx.x) * 8;
  float4 f0 = *(const float4*)(src + i);
  float4 f1 = *(const float4*)(src + i + 4);
  v8bf o;
  o[0] = (__bf16)f0.x; o[1] = (__bf16)f0.y; o[2] = (__bf16)f0.z; o[3] = (__bf16)f0.w;
  o[4] = (__bf16)f1.x; o[5] = (__bf16)f1.y; o[6] = (__bf16)f1.z; o[7] = (__bf16)f1.w;
  *(v8bf*)(dst + i) = o;
}

// ---------------------------------------------------------------------------
// V [b][kn][d] fp32 -> Vt [b][d][kn] bf16, 32x32 LDS-tiled (both sides coalesced)
// ---------------------------------------------------------------------------
__global__ void transpose_cvt_v_kernel(const float* __restrict__ V,
                                       __bf16* __restrict__ Vt) {
  __shared__ float tile[32][33];
  const int tilesPerBatch = (DD / 32) * (KL / 32); // 32 * 64
  int blk = blockIdx.x;
  int b   = blk / tilesPerBatch;
  int r   = blk % tilesPerBatch;
  int d0  = (r / (KL / 32)) * 32;
  int kn0 = (r % (KL / 32)) * 32;
  int tx = threadIdx.x & 31, ty = threadIdx.x >> 5;   // 32 x 8
  const float* src = V + ((size_t)b * KL + kn0) * DD + d0;
#pragma unroll
  for (int k = 0; k < 4; ++k) {
    int i = ty + k * 8;
    tile[i][tx] = src[(size_t)i * DD + tx];
  }
  __syncthreads();
  __bf16* dst = Vt + ((size_t)b * DD + d0) * KL + kn0;
#pragma unroll
  for (int k = 0; k < 4; ++k) {
    int j = ty + k * 8;
    dst[(size_t)j * KL + tx] = (__bf16)tile[tx][j];
  }
}

// ---------------------------------------------------------------------------
// Main fused attention kernel: one workgroup = (batch, 32 q-rows).
//   Phase 1: S[32][2048] = (Q Kt)/sqrt(D)  -> LDS (raw scaled scores)
//   Phase 2: coalesced mask add + key_pad override, row softmax in LDS,
//            write attn_weights fp32, pack P to bf16 in-place (low half of row)
//   Phase 3: ctx[32][1024] = P V  (A-frags read packed bf16 via ds_load_b128)
// LDS: 32*2048 fp32 = 256 KB (CDNA5 WGP has 320 KB).
// ---------------------------------------------------------------------------
__global__ __launch_bounds__(256) void attn_main_kernel(
    const __bf16* __restrict__ Qb, const __bf16* __restrict__ Kb,
    const __bf16* __restrict__ Vt, const float* __restrict__ amask,
    const unsigned char* __restrict__ kpm,
    float* __restrict__ ctx, float* __restrict__ aw) {
  extern __shared__ float Ssh[];          // [32][KL]
  const int tid  = threadIdx.x;
  const int lane = tid & 31;
  const int w    = tid >> 5;              // wave id 0..7
  const int b    = blockIdx.x >> 6;       // 64 q-tiles per batch
  const int q0   = (blockIdx.x & 63) << 5;

  const float scale = 0.03125f;           // 1/sqrt(1024)
  const int rowA  = lane & 15;            // A-frag row within M-tile
  const int koffA = (lane >> 4) << 4;     // A-frag K base (0 or 16)
  const int koffB = (lane >> 4) << 4;     // B-frag K base (0 or 16)

  // ---------------- Phase 1: raw scaled scores -> LDS ----------------
  const __bf16* q0p = Qb + (size_t)(b * QL + q0 + rowA) * DD;
  const __bf16* q1p = q0p + (size_t)16 * DD;

  for (int t = w; t < KL / 16; t += 8) {     // each wave owns 16 kn-tiles
    const int kn0 = t << 4;
    const __bf16* kp = Kb + (size_t)(b * KL + kn0 + rowA) * DD;
    v8f acc0 = {}; v8f acc1 = {};
#pragma unroll 4
    for (int d0 = 0; d0 < DD; d0 += 32) {
      V16 a0, a1;
      a0.h[0] = *(const v8bf*)(q0p + d0 + koffA);
      a0.h[1] = *(const v8bf*)(q0p + d0 + koffA + 8);
      a1.h[0] = *(const v8bf*)(q1p + d0 + koffA);
      a1.h[1] = *(const v8bf*)(q1p + d0 + koffA + 8);
      v16bf bb = *(const v16bf*)(kp + d0 + koffB);
      acc0 = __builtin_amdgcn_wmma_f32_16x16x32_bf16(false, a0.v, false, bb,
                                                     (short)0, acc0, false, false);
      acc1 = __builtin_amdgcn_wmma_f32_16x16x32_bf16(false, a1.v, false, bb,
                                                     (short)0, acc1, false, false);
    }
    const int kn = kn0 + (lane & 15);
    const int rb = (lane >> 4) << 3;
#pragma unroll
    for (int r = 0; r < 8; ++r) {
      int row0 = rb + r;
      Ssh[row0 * KL + kn]        = acc0[r] * scale;
      Ssh[(row0 + 16) * KL + kn] = acc1[r] * scale;
    }
  }
  __syncthreads();

  // ------- Phase 2: masks (coalesced) + softmax + aw write + bf16 pack -------
  float* awb = aw + (size_t)(b * QL + q0) * KL;
  const unsigned char* kprow = kpm + (size_t)b * KL;
  for (int rr = 0; rr < 4; ++rr) {
    const int row = (w << 2) + rr;           // 8 waves * 4 rows = 32
    float* srow = Ssh + row * KL;
    const float* amrow = amask + (size_t)(q0 + row) * KL;
    // masks + max (all accesses lane-contiguous along columns)
    float mx = -3.4e38f;
    for (int c = lane; c < KL; c += 32) {
      float s = srow[c] + amrow[c];          // add attn_mask first
      s = kprow[c] ? NEG_BIG : s;            // then key_pad override
      srow[c] = s;
      mx = fmaxf(mx, s);
    }
#pragma unroll
    for (int off = 16; off; off >>= 1) mx = fmaxf(mx, __shfl_xor(mx, off));
    float sum = 0.f;
    for (int c = lane; c < KL; c += 32) {
      float e = __expf(srow[c] - mx);
      srow[c] = e;
      sum += e;
    }
#pragma unroll
    for (int off = 16; off; off >>= 1) sum += __shfl_xor(sum, off);
    const float inv = 1.0f / sum;
    float* arow = awb + (size_t)row * KL;
    // Each lane owns column pair (2*lane, 2*lane+1), stride 64.
    // Safe in-place pack: iter k reads dwords [64k,64k+64), writes [32k,32k+32);
    // per-lane load precedes store (data dep), future reads are past all writes.
    for (int k = 0; k < KL / 64; ++k) {
      int c = 2 * lane + 64 * k;
      float2 e2 = *(const float2*)(srow + c);      // ds_load_b64
      float p0 = e2.x * inv, p1 = e2.y * inv;
      float2 o; o.x = p0; o.y = p1;
      *(float2*)(arow + c) = o;                    // global_store_b64
      BF2 pk; pk.h[0] = (__bf16)p0; pk.h[1] = (__bf16)p1;
      ((unsigned*)srow)[c >> 1] = pk.u;            // ds_store_b32 (packed bf16)
    }
  }
  __syncthreads();

  // ---------------- Phase 3: ctx = P @ V ----------------
  // Wave w owns d-columns [w*128, w*128+128): 2 M-tiles x 8 N-tiles accum.
  v8f acc[2][8];
#pragma unroll
  for (int m = 0; m < 2; ++m)
#pragma unroll
    for (int n = 0; n < 8; ++n) acc[m][n] = (v8f){};

  const __bf16* vbase = Vt + (size_t)b * DD * KL + (size_t)(w * 128) * KL;
  const int dlane = lane & 15;
  const __bf16* prow0 = (const __bf16*)(Ssh + rowA * KL);        // packed P row
  const __bf16* prow1 = (const __bf16*)(Ssh + (rowA + 16) * KL);

  for (int kn0 = 0; kn0 < KL; kn0 += 32) {
    V16 a0, a1;
    a0.h[0] = *(const v8bf*)(prow0 + kn0 + koffA);       // ds_load_b128
    a0.h[1] = *(const v8bf*)(prow0 + kn0 + koffA + 8);
    a1.h[0] = *(const v8bf*)(prow1 + kn0 + koffA);
    a1.h[1] = *(const v8bf*)(prow1 + kn0 + koffA + 8);
#pragma unroll
    for (int nt = 0; nt < 8; ++nt) {
      const __bf16* vp = vbase + (size_t)(nt * 16 + dlane) * KL + kn0 + koffB;
      v16bf bb = *(const v16bf*)vp;
      acc[0][nt] = __builtin_amdgcn_wmma_f32_16x16x32_bf16(false, a0.v, false, bb,
                                                           (short)0, acc[0][nt], false, false);
      acc[1][nt] = __builtin_amdgcn_wmma_f32_16x16x32_bf16(false, a1.v, false, bb,
                                                           (short)0, acc[1][nt], false, false);
    }
  }

  float* cbase = ctx + (size_t)(b * QL + q0) * DD + w * 128;
  const int rb2 = (lane >> 4) << 3;
#pragma unroll
  for (int m = 0; m < 2; ++m)
#pragma unroll
    for (int nt = 0; nt < 8; ++nt)
#pragma unroll
      for (int r = 0; r < 8; ++r) {
        int row = m * 16 + rb2 + r;
        cbase[(size_t)row * DD + nt * 16 + dlane] = acc[m][nt][r];
      }
}

// ---------------------------------------------------------------------------
extern "C" void kernel_launch(void* const* d_in, const int* in_sizes, int n_in,
                              void* d_out, int out_size, void* d_ws, size_t ws_size,
                              hipStream_t stream) {
  const float* Q = (const float*)d_in[0];
  const float* K = (const float*)d_in[1];
  const float* V = (const float*)d_in[2];
  const float* amask = (const float*)d_in[3];
  const unsigned char* kpm = (const unsigned char*)d_in[4]; // jnp.bool_ = 1 byte

  float* ctx = (float*)d_out;                               // [B, QL, D]
  float* aw  = (float*)d_out + (size_t)BB * QL * DD;        // [B, QL, KL]

  const size_t nQ = (size_t)BB * QL * DD;                   // == n for K, V
  __bf16* Qb = (__bf16*)d_ws;
  __bf16* Kb = Qb + nQ;
  __bf16* Vt = Kb + nQ;

  // fp32 -> bf16 converts (streamed, memory-bound)
  cvt_bf16_kernel<<<(unsigned)(nQ / (256 * 8)), 256, 0, stream>>>(Q, Qb);
  cvt_bf16_kernel<<<(unsigned)(nQ / (256 * 8)), 256, 0, stream>>>(K, Kb);
  transpose_cvt_v_kernel<<<BB * (DD / 32) * (KL / 32), 256, 0, stream>>>(V, Vt);

  // 256 KB dynamic LDS for the score strip (CDNA5 WGP: 320 KB)
  const int ldsBytes = 32 * KL * (int)sizeof(float);
  (void)hipFuncSetAttribute((const void*)attn_main_kernel,
                            hipFuncAttributeMaxDynamicSharedMemorySize, ldsBytes);

  attn_main_kernel<<<BB * (QL / 32), 256, ldsBytes, stream>>>(
      Qb, Kb, Vt, amask, kpm, ctx, aw);
}
